// CausalSelfAttention_35837207118564
// MI455X (gfx1250) — matrix operand order repaired and checked by reading, not compiled, not run on hardware
//
#include <hip/hip_runtime.h>

typedef __bf16 bf16_t;
typedef bf16_t v16bf __attribute__((ext_vector_type(16)));
typedef bf16_t v2bf  __attribute__((ext_vector_type(2)));
typedef float  v8f  __attribute__((ext_vector_type(8)));

#define DEVFN static __device__ __forceinline__

constexpr int kB = 2, kT = 1024, kC = 768, kH = 12, kHD = 64;
constexpr int kN3C = 3 * kC;                 // 2304
constexpr float kQScale = 0.125f;            // 1/sqrt(64)
constexpr float kNegInf = -3.0e38f;

#if defined(__has_builtin)
#if __has_builtin(__builtin_amdgcn_cvt_pk_bf16_f32)
#define HAVE_PK_BF16 1
#endif
#endif
#ifndef HAVE_PK_BF16
#define HAVE_PK_BF16 0
#endif

DEVFN bf16_t f2bf(float f) {
  unsigned u = __builtin_bit_cast(unsigned, f);
  u += 0x7FFFu + ((u >> 16) & 1u);           // round-to-nearest-even
  unsigned short h = (unsigned short)(u >> 16);
  return __builtin_bit_cast(bf16_t, h);
}

// Pack two floats into one bf16x2 dword (lo -> [15:0], hi -> [31:16]).
DEVFN unsigned pkbf2(float lo, float hi) {
#if HAVE_PK_BF16
  v2bf r = __builtin_amdgcn_cvt_pk_bf16_f32(lo, hi);
  return __builtin_bit_cast(unsigned, r);
#else
  unsigned a = __builtin_bit_cast(unsigned, lo);
  a += 0x7FFFu + ((a >> 16) & 1u);
  unsigned b = __builtin_bit_cast(unsigned, hi);
  b += 0x7FFFu + ((b >> 16) & 1u);
  return (b & 0xFFFF0000u) | (a >> 16);
#endif
}

DEVFN v8f wmma_bf16(v16bf a, v16bf b, v8f c) {
  return __builtin_amdgcn_wmma_f32_16x16x32_bf16(false, a, false, b, (short)0, c,
                                                 false, false);
}

// ---------------------------------------------------------------------------
// Fragment loaders: per-lane data is contiguous -> 2 x b128 loads each.
// ---------------------------------------------------------------------------

// A fragment (16x32 bf16) from row-major [16][rs] tile (rs in elements).
DEVFN v16bf load_a_frag(const bf16_t* t, int rs, int lane) {
  const int m = lane & 15, hi = lane >> 4;
  const bf16_t* p = t + m * rs + 8 * hi;
  union { uint4 q[2]; v16bf v; } u;
  u.q[0] = *(const uint4*)(p);        // K = 8hi .. 8hi+7
  u.q[1] = *(const uint4*)(p + 16);   // K = 16+8hi .. 16+8hi+7
  return u.v;
}

// B fragment (32x16 bf16) from an n-major source: element (k,n) at t[n*rs + k].
DEVFN v16bf load_b_fragT(const bf16_t* t, int rs, int lane) {
  const int n = lane & 15, hi = lane >> 4;
  const bf16_t* p = t + n * rs + 8 * hi;
  union { uint4 q[2]; v16bf v; } u;
  u.q[0] = *(const uint4*)(p);        // K = 8hi .. 8hi+7
  u.q[1] = *(const uint4*)(p + 16);   // K = 16+8hi .. 16+8hi+7
  return u.v;
}

// ---------------------------------------------------------------------------
// Tile staging helpers (each thread owns 16 contiguous bf16 = 32 bytes).
// ---------------------------------------------------------------------------

// Async VMEM->LDS copy of 32 bytes per lane (CDNA5 GLOBAL_LOAD_ASYNC_TO_LDS).
// IOFFSET advances both the LDS and global addresses (ISA 10.7/15.18).
DEVFN void async_copy16_bf(bf16_t* lds_dst, const bf16_t* gsrc) {
  unsigned loff =
      (unsigned)(size_t)(__attribute__((address_space(3))) bf16_t*)lds_dst;
  asm volatile(
      "global_load_async_to_lds_b128 %0, %1, off\n\t"
      "global_load_async_to_lds_b128 %0, %1, off offset:16"
      :
      : "v"(loff), "v"(gsrc)
      : "memory");
}
DEVFN void async_wait0() {
  asm volatile("s_wait_asynccnt 0x0" ::: "memory");
}

DEVFN float redmax16(float v) {
#pragma unroll
  for (int o = 1; o < 16; o <<= 1) v = fmaxf(v, __shfl_xor(v, o, 32));
  return v;
}
DEVFN float redsum16(float v) {
#pragma unroll
  for (int o = 1; o < 16; o <<= 1) v += __shfl_xor(v, o, 32);
  return v;
}

// ---------------------------------------------------------------------------
// Prep kernels: one-shot f32 -> bf16 conversions (hoisted out of the GEMMs so
// tiles are never re-converted per block).
// ---------------------------------------------------------------------------
__global__ __launch_bounds__(256) void cvt_bf16_kernel(
    const float* __restrict__ src, bf16_t* __restrict__ dst, int n) {
  const int i = (blockIdx.x * 256 + threadIdx.x) * 8;
  if (i >= n) return;
  const float4 f0 = *(const float4*)(src + i);
  const float4 f1 = *(const float4*)(src + i + 4);
  uint4 q;
  q.x = pkbf2(f0.x, f0.y); q.y = pkbf2(f0.z, f0.w);
  q.z = pkbf2(f1.x, f1.y); q.w = pkbf2(f1.z, f1.w);
  *(uint4*)(dst + i) = q;
}

__global__ __launch_bounds__(256) void cvt_w3_kernel(
    const float* __restrict__ src, bf16_t* __restrict__ dW,
    bf16_t* __restrict__ dP, bf16_t* __restrict__ dN, int n) {
  const int i = (blockIdx.x * 256 + threadIdx.x) * 8;
  if (i >= n) return;
  float f[8];
  *(float4*)&f[0] = *(const float4*)(src + i);
  *(float4*)&f[4] = *(const float4*)(src + i + 4);
  uint4 qw, qp, qn;
#pragma unroll
  for (int j = 0; j < 4; ++j) {
    const float a = f[2 * j], b = f[2 * j + 1];
    (&qw.x)[j] = pkbf2(a, b);
    (&qp.x)[j] = pkbf2(fmaxf(a, 0.f), fmaxf(b, 0.f));
    (&qn.x)[j] = pkbf2(fminf(a, 0.f), fminf(b, 0.f));
  }
  *(uint4*)(dW + i) = qw;
  *(uint4*)(dP + i) = qp;
  *(uint4*)(dN + i) = qn;
}

// ---------------------------------------------------------------------------
// Kernel 1: fused QKV + interval-bound projections (all operands bf16).
// A fragments load straight from global; W/W+/W- tiles async-staged in LDS.
// q/k tensors land as [B*H][T][HD]; v tensors land d-major as [B*H][HD][T].
// ---------------------------------------------------------------------------
__global__ __launch_bounds__(128) void qkv_bounds_gemm(
    const bf16_t* __restrict__ xb, const bf16_t* __restrict__ xlb,
    const bf16_t* __restrict__ xub, const bf16_t* __restrict__ Wab,
    const bf16_t* __restrict__ Wpb, const bf16_t* __restrict__ Wnb,
    bf16_t* __restrict__ qkv_base) {
  __shared__ __align__(16) bf16_t Bs[3][64 * 32];  // W / W+ / W-  n-major
  const int tid = threadIdx.x, lane = tid & 31, w = tid >> 5;
  const int hi = lane >> 4, nlo = lane & 15;
  const int m0 = blockIdx.x * 64, n0 = blockIdx.y * 64;
  const int tt16 = tid * 16;
  const int cr = tt16 >> 5, cc = tt16 & 31;        // copy row / col-half

  const v8f z = {0.f, 0.f, 0.f, 0.f, 0.f, 0.f, 0.f, 0.f};
  v8f acc[3][4];
#pragma unroll
  for (int s = 0; s < 3; ++s)
#pragma unroll
    for (int nt = 0; nt < 4; ++nt) acc[s][nt] = z;

  const size_t aRow = (size_t)(m0 + 16 * w) * kC;
  for (int k0 = 0; k0 < kC; k0 += 32) {
    __syncthreads();
    {
      const size_t gw = (size_t)(n0 + cr) * kC + k0 + cc;
      async_copy16_bf(&Bs[0][tt16], Wab + gw);
      async_copy16_bf(&Bs[1][tt16], Wpb + gw);
      async_copy16_bf(&Bs[2][tt16], Wnb + gw);
    }
    async_wait0();
    __syncthreads();

    const v16bf aX = load_a_frag(xb + aRow + k0, kC, lane);
    const v16bf aL = load_a_frag(xlb + aRow + k0, kC, lane);
    const v16bf aU = load_a_frag(xub + aRow + k0, kC, lane);
#pragma unroll
    for (int nt = 0; nt < 4; ++nt) {
      const v16bf bW = load_b_fragT(&Bs[0][nt * 16 * 32], 32, lane);
      const v16bf bP = load_b_fragT(&Bs[1][nt * 16 * 32], 32, lane);
      const v16bf bN = load_b_fragT(&Bs[2][nt * 16 * 32], 32, lane);
      acc[0][nt] = wmma_bf16(aX, bW, acc[0][nt]);
      acc[1][nt] = wmma_bf16(aL, bP, acc[1][nt]);
      acc[1][nt] = wmma_bf16(aU, bN, acc[1][nt]);
      acc[2][nt] = wmma_bf16(aU, bP, acc[2][nt]);
      acc[2][nt] = wmma_bf16(aL, bN, acc[2][nt]);
    }
  }

  const size_t nQKV = (size_t)kB * kH * kT * kHD;
#pragma unroll
  for (int s = 0; s < 3; ++s)
#pragma unroll
    for (int nt = 0; nt < 4; ++nt)
#pragma unroll
      for (int r = 0; r < 8; ++r) {
        const int row = m0 + w * 16 + r + 8 * hi;   // global M (b*T + t)
        const int col = n0 + nt * 16 + nlo;         // global N in [0, 3C)
        const int bb = row >> 10, tt = row & (kT - 1);
        const int sel = col / kC, c = col - sel * kC;
        const int hh = c >> 6, dd = c & 63;
        float v = acc[s][nt][r];
        if (sel == 0) v *= kQScale;
        const size_t head = (size_t)(bb * kH + hh) * (kT * kHD);
        const size_t off = (sel == 2) ? ((size_t)dd * kT + tt)    // v: d-major
                                      : ((size_t)tt * kHD + dd);  // q,k: t-major
        qkv_base[(size_t)(s * 3 + sel) * nQKV + head + off] = f2bf(v);
      }
}

// ---------------------------------------------------------------------------
// Kernel 2: main-path causal flash attention. One wave = 16 query rows.
// Q,K: [B*H][T][HD]; V: [B*H][HD][T] (d-major). Async VMEM->LDS staging.
// ---------------------------------------------------------------------------
__global__ __launch_bounds__(128) void attn_main(
    const bf16_t* __restrict__ Q, const bf16_t* __restrict__ K,
    const bf16_t* __restrict__ V, bf16_t* __restrict__ Y) {
  __shared__ __align__(16) bf16_t Ks[32 * 64];     // [key][d]
  __shared__ __align__(16) bf16_t VsT[64 * 32];    // [d][key]
  __shared__ __align__(16) bf16_t Ps[4][16 * 32];
  const int tid = threadIdx.x, lane = tid & 31, w = tid >> 5;
  const int hi = lane >> 4, nlo = lane & 15;
  const int bh = blockIdx.y, q0 = blockIdx.x * 64, qw0 = q0 + 16 * w;
  const bf16_t* Qb = Q + (size_t)bh * kT * kHD;
  const bf16_t* Kb = K + (size_t)bh * kT * kHD;
  const bf16_t* Vb = V + (size_t)bh * kT * kHD;
  const int tt16 = tid * 16;
  const int vd = tt16 >> 5, vk = tt16 & 31;        // V copy: fixed d, 16 keys

  v16bf qf[2];
#pragma unroll
  for (int c2 = 0; c2 < 2; ++c2)
    qf[c2] = load_a_frag(Qb + (size_t)qw0 * kHD + 32 * c2, kHD, lane);

  const v8f z = {0.f, 0.f, 0.f, 0.f, 0.f, 0.f, 0.f, 0.f};
  v8f acc[4] = {z, z, z, z};
  float mr[8], lr[8];
#pragma unroll
  for (int r = 0; r < 8; ++r) { mr[r] = kNegInf; lr[r] = 0.f; }

  const int nk = (q0 + 64) >> 5;
  for (int kt = 0; kt < nk; ++kt) {
    const int kb0 = kt * 32;
    __syncthreads();
    async_copy16_bf(&Ks[tt16], Kb + (size_t)kb0 * kHD + tt16);
    async_copy16_bf(&VsT[tt16], Vb + (size_t)vd * kT + kb0 + vk);
    if (kt + 1 < nk) {                              // prefetch next tiles
      __builtin_prefetch(Kb + (size_t)(kb0 + 32) * kHD + tt16, 0, 1);
      __builtin_prefetch(Vb + (size_t)vd * kT + kb0 + 32 + vk, 0, 1);
    }
    async_wait0();
    __syncthreads();
    if (kb0 <= qw0 + 15) {                          // wave-uniform, EXEC full
      v8f S0 = z, S1 = z;
      S0 = wmma_bf16(qf[0], load_b_fragT(&Ks[0], 64, lane), S0);
      S0 = wmma_bf16(qf[1], load_b_fragT(&Ks[32], 64, lane), S0);
      S1 = wmma_bf16(qf[0], load_b_fragT(&Ks[16 * 64], 64, lane), S1);
      S1 = wmma_bf16(qf[1], load_b_fragT(&Ks[16 * 64 + 32], 64, lane), S1);
      const bool diag = (kb0 + 31 > qw0);           // wave-uniform
#pragma unroll
      for (int r = 0; r < 8; ++r) {
        if (diag) {
          const int qg = qw0 + r + 8 * hi;
          if (kb0 + nlo > qg) S0[r] = kNegInf;
          if (kb0 + 16 + nlo > qg) S1[r] = kNegInf;
        }
        const float mx = redmax16(fmaxf(S0[r], S1[r]));
        const float mn = fmaxf(mr[r], mx);
        const float alpha = __expf(mr[r] - mn);
        const float p0 = __expf(S0[r] - mn), p1 = __expf(S1[r] - mn);
        lr[r] = lr[r] * alpha + redsum16(p0 + p1);
        mr[r] = mn;
#pragma unroll
        for (int o = 0; o < 4; ++o) acc[o][r] = acc[o][r] * alpha;
        Ps[w][(r + 8 * hi) * 32 + nlo] = f2bf(p0);
        Ps[w][(r + 8 * hi) * 32 + 16 + nlo] = f2bf(p1);
      }
      const v16bf pa = load_a_frag(&Ps[w][0], 32, lane);  // same-wave DS in-order
#pragma unroll
      for (int o = 0; o < 4; ++o)
        acc[o] = wmma_bf16(pa, load_b_fragT(&VsT[o * 16 * 32], 32, lane), acc[o]);
    }
  }

  const int bb = bh / kH, hh = bh % kH;
#pragma unroll
  for (int o = 0; o < 4; ++o)
#pragma unroll
    for (int r = 0; r < 8; ++r) {
      const int tt = qw0 + r + 8 * hi;
      const float v = acc[o][r] / lr[r];
      Y[((size_t)(bb * kT + tt)) * kC + hh * kHD + o * 16 + nlo] = f2bf(v);
    }
}

// ---------------------------------------------------------------------------
// Kernel 3: bound attention. 4 (q,k) combos in-kernel; each combo drives two
// accumulators (vL / vU); 8 candidates min/max-merged in registers.
// ---------------------------------------------------------------------------
__global__ __launch_bounds__(128) void attn_bounds(
    const bf16_t* __restrict__ qLp, const bf16_t* __restrict__ qUp,
    const bf16_t* __restrict__ kLp, const bf16_t* __restrict__ kUp,
    const bf16_t* __restrict__ vLp, const bf16_t* __restrict__ vUp,
    bf16_t* __restrict__ Ymin, bf16_t* __restrict__ Ymax) {
  __shared__ __align__(16) bf16_t Ks[32 * 64];     // [key][d]
  __shared__ __align__(16) bf16_t VsL[64 * 32];    // [d][key]
  __shared__ __align__(16) bf16_t VsU[64 * 32];    // [d][key]
  __shared__ __align__(16) bf16_t Ps[4][16 * 32];
  const int tid = threadIdx.x, lane = tid & 31, w = tid >> 5;
  const int hi = lane >> 4, nlo = lane & 15;
  const int bh = blockIdx.y, q0 = blockIdx.x * 64, qw0 = q0 + 16 * w;
  const size_t base = (size_t)bh * kT * kHD;
  const bf16_t* vLb = vLp + base;
  const bf16_t* vUb = vUp + base;
  const int tt16 = tid * 16;
  const int vd = tt16 >> 5, vk = tt16 & 31;

  v16bf qfL[2], qfU[2];
#pragma unroll
  for (int c2 = 0; c2 < 2; ++c2) {
    qfL[c2] = load_a_frag(qLp + base + (size_t)qw0 * kHD + 32 * c2, kHD, lane);
    qfU[c2] = load_a_frag(qUp + base + (size_t)qw0 * kHD + 32 * c2, kHD, lane);
  }

  const v8f z = {0.f, 0.f, 0.f, 0.f, 0.f, 0.f, 0.f, 0.f};
  v8f ymin[4], ymax[4];
#pragma unroll
  for (int o = 0; o < 4; ++o)
#pragma unroll
    for (int r = 0; r < 8; ++r) { ymin[o][r] = 3.0e38f; ymax[o][r] = kNegInf; }

  const int nk = (q0 + 64) >> 5;
  for (int qs = 0; qs < 2; ++qs) {
    const v16bf* qf = qs ? qfU : qfL;
    for (int ks = 0; ks < 2; ++ks) {
      const bf16_t* Kb = (ks ? kUp : kLp) + base;
      v8f accL[4] = {z, z, z, z}, accU[4] = {z, z, z, z};
      float mr[8], lr[8];
#pragma unroll
      for (int r = 0; r < 8; ++r) { mr[r] = kNegInf; lr[r] = 0.f; }

      for (int kt = 0; kt < nk; ++kt) {
        const int kb0 = kt * 32;
        __syncthreads();
        async_copy16_bf(&Ks[tt16], Kb + (size_t)kb0 * kHD + tt16);
        async_copy16_bf(&VsL[tt16], vLb + (size_t)vd * kT + kb0 + vk);
        async_copy16_bf(&VsU[tt16], vUb + (size_t)vd * kT + kb0 + vk);
        if (kt + 1 < nk) {
          __builtin_prefetch(Kb + (size_t)(kb0 + 32) * kHD + tt16, 0, 1);
          __builtin_prefetch(vLb + (size_t)vd * kT + kb0 + 32 + vk, 0, 1);
          __builtin_prefetch(vUb + (size_t)vd * kT + kb0 + 32 + vk, 0, 1);
        }
        async_wait0();
        __syncthreads();
        if (kb0 <= qw0 + 15) {
          v8f S0 = z, S1 = z;
          S0 = wmma_bf16(qf[0], load_b_fragT(&Ks[0], 64, lane), S0);
          S0 = wmma_bf16(qf[1], load_b_fragT(&Ks[32], 64, lane), S0);
          S1 = wmma_bf16(qf[0], load_b_fragT(&Ks[16 * 64], 64, lane), S1);
          S1 = wmma_bf16(qf[1], load_b_fragT(&Ks[16 * 64 + 32], 64, lane), S1);
          const bool diag = (kb0 + 31 > qw0);       // wave-uniform
#pragma unroll
          for (int r = 0; r < 8; ++r) {
            if (diag) {
              const int qg = qw0 + r + 8 * hi;
              if (kb0 + nlo > qg) S0[r] = kNegInf;
              if (kb0 + 16 + nlo > qg) S1[r] = kNegInf;
            }
            const float mx = redmax16(fmaxf(S0[r], S1[r]));
            const float mn = fmaxf(mr[r], mx);
            const float alpha = __expf(mr[r] - mn);
            const float p0 = __expf(S0[r] - mn), p1 = __expf(S1[r] - mn);
            lr[r] = lr[r] * alpha + redsum16(p0 + p1);
            mr[r] = mn;
#pragma unroll
            for (int o = 0; o < 4; ++o) {
              accL[o][r] = accL[o][r] * alpha;
              accU[o][r] = accU[o][r] * alpha;
            }
            Ps[w][(r + 8 * hi) * 32 + nlo] = f2bf(p0);
            Ps[w][(r + 8 * hi) * 32 + 16 + nlo] = f2bf(p1);
          }
          const v16bf pa = load_a_frag(&Ps[w][0], 32, lane);
#pragma unroll
          for (int o = 0; o < 4; ++o) {
            const v16bf bL = load_b_fragT(&VsL[o * 16 * 32], 32, lane);
            const v16bf bU = load_b_fragT(&VsU[o * 16 * 32], 32, lane);
            accL[o] = wmma_bf16(pa, bL, accL[o]);
            accU[o] = wmma_bf16(pa, bU, accU[o]);
          }
        }
      }
#pragma unroll
      for (int o = 0; o < 4; ++o)
#pragma unroll
        for (int r = 0; r < 8; ++r) {
          const float inv = 1.f / lr[r];
          const float cL = accL[o][r] * inv, cU = accU[o][r] * inv;
          ymin[o][r] = fminf(ymin[o][r], fminf(cL, cU));
          ymax[o][r] = fmaxf(ymax[o][r], fmaxf(cL, cU));
        }
    }
  }

  const int bb = bh / kH, hh = bh % kH;
#pragma unroll
  for (int o = 0; o < 4; ++o)
#pragma unroll
    for (int r = 0; r < 8; ++r) {
      const int tt = qw0 + r + 8 * hi;
      const size_t oi = ((size_t)(bb * kT + tt)) * kC + hh * kHD + o * 16 + nlo;
      Ymin[oi] = f2bf(ymin[o][r]);
      Ymax[oi] = f2bf(ymax[o][r]);
    }
}

// ---------------------------------------------------------------------------
// Kernel 4: output projections; y/y_min/y_max share one async-staged W tile;
// A fragments load straight from global bf16.
// ---------------------------------------------------------------------------
__global__ __launch_bounds__(128) void proj_gemm(
    const bf16_t* __restrict__ Y0, const bf16_t* __restrict__ Y1,
    const bf16_t* __restrict__ Y2, const bf16_t* __restrict__ Wpb16,
    float* __restrict__ out) {
  __shared__ __align__(16) bf16_t Bs[64 * 32];     // n-major (64 n x 32 k)
  const int tid = threadIdx.x, lane = tid & 31, w = tid >> 5;
  const int hi = lane >> 4, nlo = lane & 15;
  const int m0 = blockIdx.x * 64, n0 = blockIdx.y * 64;
  const int tt16 = tid * 16;
  const int cr = tt16 >> 5, cc = tt16 & 31;

  const v8f z = {0.f, 0.f, 0.f, 0.f, 0.f, 0.f, 0.f, 0.f};
  v8f acc[3][4];
#pragma unroll
  for (int s = 0; s < 3; ++s)
#pragma unroll
    for (int nt = 0; nt < 4; ++nt) acc[s][nt] = z;

  const size_t aRow = (size_t)(m0 + 16 * w) * kC;
  for (int k0 = 0; k0 < kC; k0 += 32) {
    __syncthreads();
    async_copy16_bf(&Bs[tt16], Wpb16 + (size_t)(n0 + cr) * kC + k0 + cc);
    async_wait0();
    __syncthreads();
    const v16bf a0 = load_a_frag(Y0 + aRow + k0, kC, lane);
    const v16bf a1 = load_a_frag(Y1 + aRow + k0, kC, lane);
    const v16bf a2 = load_a_frag(Y2 + aRow + k0, kC, lane);
#pragma unroll
    for (int nt = 0; nt < 4; ++nt) {
      const v16bf bfrag = load_b_fragT(&Bs[nt * 16 * 32], 32, lane);
      acc[0][nt] = wmma_bf16(a0, bfrag, acc[0][nt]);
      acc[1][nt] = wmma_bf16(a1, bfrag, acc[1][nt]);
      acc[2][nt] = wmma_bf16(a2, bfrag, acc[2][nt]);
    }
  }

  const size_t nBTC = (size_t)kB * kT * kC;
#pragma unroll
  for (int s = 0; s < 3; ++s)
#pragma unroll
    for (int nt = 0; nt < 4; ++nt)
#pragma unroll
      for (int r = 0; r < 8; ++r) {
        const int row = m0 + w * 16 + r + 8 * hi;
        const int col = n0 + nt * 16 + nlo;
        out[(size_t)s * nBTC + (size_t)row * kC + col] = acc[s][nt][r];
      }
}

extern "C" void kernel_launch(void* const* d_in, const int* in_sizes, int n_in,
                              void* d_out, int out_size, void* d_ws,
                              size_t ws_size, hipStream_t stream) {
  (void)in_sizes; (void)n_in; (void)out_size; (void)ws_size;
  const float* x  = (const float*)d_in[0];
  const float* xl = (const float*)d_in[1];
  const float* xu = (const float*)d_in[2];
  const float* Wa = (const float*)d_in[3];
  const float* Wp = (const float*)d_in[4];
  float* out = (float*)d_out;

  const size_t nQKV = (size_t)kB * kH * kT * kHD;   // == B*T*C = 1,572,864
  const size_t nWA  = (size_t)kN3C * kC;            // 1,769,472
  const size_t nWP  = (size_t)kC * kC;              // 589,824
  bf16_t* ws = (bf16_t*)d_ws;
  bf16_t* qkv_base = ws;                   // 9 tensors: q,k,v|qL,kL,vL|qU,kU,vU
  bf16_t* ya    = ws + 9 * nQKV;           // main attention out, [B*T][C]
  bf16_t* ymin  = ws + 10 * nQKV;
  bf16_t* ymax  = ws + 11 * nQKV;
  bf16_t* xb    = ws + 12 * nQKV;
  bf16_t* xlb   = ws + 13 * nQKV;
  bf16_t* xub   = ws + 14 * nQKV;
  bf16_t* Wab   = ws + 15 * nQKV;
  bf16_t* Wpb   = Wab + nWA;
  bf16_t* Wnb   = Wab + 2 * nWA;
  bf16_t* Wpb16 = Wab + 3 * nWA;

  dim3 blk(128);
  // one-shot conversions
  cvt_bf16_kernel<<<dim3((int)(nQKV / 8 / 256)), dim3(256), 0, stream>>>(
      x, xb, (int)nQKV);
  cvt_bf16_kernel<<<dim3((int)(nQKV / 8 / 256)), dim3(256), 0, stream>>>(
      xl, xlb, (int)nQKV);
  cvt_bf16_kernel<<<dim3((int)(nQKV / 8 / 256)), dim3(256), 0, stream>>>(
      xu, xub, (int)nQKV);
  cvt_w3_kernel<<<dim3((int)(nWA / 8 / 256)), dim3(256), 0, stream>>>(
      Wa, Wab, Wpb, Wnb, (int)nWA);
  cvt_bf16_kernel<<<dim3((int)(nWP / 8 / 256)), dim3(256), 0, stream>>>(
      Wp, Wpb16, (int)nWP);

  qkv_bounds_gemm<<<dim3((kB * kT) / 64, kN3C / 64), blk, 0, stream>>>(
      xb, xlb, xub, Wab, Wpb, Wnb, qkv_base);
  attn_main<<<dim3(kT / 64, kB * kH), blk, 0, stream>>>(
      qkv_base + 0 * nQKV, qkv_base + 1 * nQKV, qkv_base + 2 * nQKV, ya);
  attn_bounds<<<dim3(kT / 64, kB * kH), blk, 0, stream>>>(
      qkv_base + 3 * nQKV, qkv_base + 6 * nQKV,   // qL, qU
      qkv_base + 4 * nQKV, qkv_base + 7 * nQKV,   // kL, kU
      qkv_base + 5 * nQKV, qkv_base + 8 * nQKV,   // vL, vU
      ymin, ymax);
  proj_gemm<<<dim3((kB * kT) / 64, kC / 64), blk, 0, stream>>>(
      ya, ymin, ymax, Wpb16, out);
}
